// MptAttention_19104014532863
// MI455X (gfx1250) — matrix operand order repaired and checked by reading, compiled, not run on hardware
//
#include <hip/hip_runtime.h>
#include <hip/hip_bf16.h>
#include <math.h>
#include <stdint.h>

typedef __attribute__((ext_vector_type(16))) __bf16 v16bf;
typedef __attribute__((ext_vector_type(8)))  __bf16 v8bf;
typedef __attribute__((ext_vector_type(8)))  float  v8f;

#define WMMA_BF16(A_, B_, C_) \
  __builtin_amdgcn_wmma_f32_16x16x32_bf16(false, (A_), false, (B_), (short)0, (C_), false, false)

constexpr int Bb = 4, S = 2048, D = 2048, H = 16, HD = 128;
constexpr int M1 = Bb * S;   // 8192 rows
constexpr int K1 = D;        // 2048
constexpr int N1 = 3 * D;    // 6144

__device__ inline v16bf cat16(v8bf a, v8bf b) {
  return __builtin_shufflevector(a, b, 0,1,2,3,4,5,6,7,8,9,10,11,12,13,14,15);
}

// low 32 bits of a flat shared pointer = LDS byte offset (ISA §10.2 aperture map)
__device__ inline unsigned lds_off_u32(const void* p) {
  return (unsigned)(unsigned long long)(uintptr_t)p;
}

// CDNA5 async global->LDS copy, tracked by ASYNCcnt (ISA §10 / 15.18.3 op 98)
__device__ inline void async_copy_b128(unsigned ldsOff, unsigned long long gaddr) {
  asm volatile("global_load_async_to_lds_b128 %0, %1, off"
               :: "v"(ldsOff), "v"(gaddr) : "memory");
}
__device__ inline void wait_asynccnt_le4() {
#if __has_builtin(__builtin_amdgcn_s_wait_asynccnt)
  __builtin_amdgcn_s_wait_asynccnt(4);
#else
  asm volatile("s_wait_asynccnt 4" ::: "memory");
#endif
}
__device__ inline void wait_asynccnt_le0() {
#if __has_builtin(__builtin_amdgcn_s_wait_asynccnt)
  __builtin_amdgcn_s_wait_asynccnt(0);
#else
  asm volatile("s_wait_asynccnt 0" ::: "memory");
#endif
}

// ---------------- precision conversion ----------------
__global__ void cvt_bf16_kernel(const float* __restrict__ in, __bf16* __restrict__ out, int n) {
  int i = blockIdx.x * blockDim.x + threadIdx.x;
  if (i < n) out[i] = (__bf16)in[i];
}

// in [K][N] fp32 -> out [N][K] bf16, 32x32 LDS-tiled (coalesced both sides). K,N % 32 == 0.
__global__ void __launch_bounds__(256)
transpose_bf16_kernel(const float* __restrict__ in, __bf16* __restrict__ out, int K, int N) {
  __shared__ __bf16 tile[32][33];
  const int n0 = blockIdx.x * 32, k0 = blockIdx.y * 32;
  const int tx = threadIdx.x & 31, ty = threadIdx.x >> 5;  // 32 x 8
#pragma unroll
  for (int i = 0; i < 32; i += 8)
    tile[ty + i][tx] = (__bf16)in[(long long)(k0 + ty + i) * N + n0 + tx];
  __syncthreads();
#pragma unroll
  for (int i = 0; i < 32; i += 8)
    out[(long long)(n0 + ty + i) * K + k0 + tx] = tile[tx][ty + i];
}

// ---------------- LDS-staged 128x128-tile bf16 GEMM (K = 2048) ----------------
// A [*,2048] bf16 row-major; Bt [N,2048] bf16 row-major (weights pre-transposed).
// 256 threads = 8 waves; wave (wm 0..3, wn 0..1) computes a 32x64 sub-tile.
// Double-buffered LDS slabs filled with global_load_async_to_lds_b128.
// MODE 0: QKV epilogue -> Q[b,h,s,d]*scale, K[b,h,s,d], V^T[b,h,d,s] (bf16)
// MODE 1: fp32 output with bias.
template <int MODE>
__global__ void __launch_bounds__(256)
gemm_bf16_kernel(const __bf16* __restrict__ A, const __bf16* __restrict__ Bt,
                 const float* __restrict__ bias,
                 __bf16* __restrict__ Qh, __bf16* __restrict__ Kh, __bf16* __restrict__ Vt,
                 float* __restrict__ FO) {
  constexpr int PITCH = 40;  // 32 + 8 pad: keeps 16B alignment, spreads banks
  __shared__ __align__(16) __bf16 Al[2][128 * PITCH];
  __shared__ __align__(16) __bf16 Bl[2][128 * PITCH];

  const int tid = threadIdx.x;
  const int wave = tid >> 5, lane = tid & 31;
  const int half = lane >> 4, lm = lane & 15;
  const int wm = wave >> 1, wn = wave & 1;
  const int mBlock = blockIdx.y * 128, nBlock = blockIdx.x * 128;

  auto fill = [&](int buf, int kk) {
#pragma unroll
    for (int i = 0; i < 2; ++i) {
      int idx = tid + i * 256;           // 512 x 16B chunks per slab
      int row = idx >> 2, ch = idx & 3;  // 128 rows x 4 chunks of 8 bf16
      async_copy_b128(lds_off_u32(&Al[buf][row * PITCH + ch * 8]),
                      (unsigned long long)(uintptr_t)(A + (long long)(mBlock + row) * K1 + kk + ch * 8));
      async_copy_b128(lds_off_u32(&Bl[buf][row * PITCH + ch * 8]),
                      (unsigned long long)(uintptr_t)(Bt + (long long)(nBlock + row) * K1 + kk + ch * 8));
    }
  };

  v8f acc[2][4];
#pragma unroll
  for (int mt = 0; mt < 2; ++mt)
#pragma unroll
    for (int nt = 0; nt < 4; ++nt) acc[mt][nt] = v8f{};

  fill(0, 0);
  for (int kk = 0; kk < K1; kk += 32) {
    const int cur = (kk >> 5) & 1;
    if (kk + 32 < K1) {
      fill(cur ^ 1, kk + 32);
      wait_asynccnt_le4();   // 4 outstanding from next slab; current slab complete
    } else {
      wait_asynccnt_le0();
    }
    __syncthreads();         // whole slab visible to all waves

    const __bf16* Ab = Al[cur];
    const __bf16* Bb = Bl[cur];
    v16bf af[2], bfr[4];
#pragma unroll
    for (int mt = 0; mt < 2; ++mt) {
      const __bf16* ap = Ab + (wm * 32 + mt * 16 + lm) * PITCH;
      af[mt] = cat16(*(const v8bf*)(ap + half * 8), *(const v8bf*)(ap + 16 + half * 8));
    }
#pragma unroll
    for (int nt = 0; nt < 4; ++nt)
      bfr[nt] = cat16(*(const v8bf*)(Bb + (wn * 64 + nt * 16 + lm) * PITCH + half * 8), *(const v8bf*)(Bb + (wn * 64 + nt * 16 + lm) * PITCH + 16 + half * 8));
#pragma unroll
    for (int mt = 0; mt < 2; ++mt)
#pragma unroll
      for (int nt = 0; nt < 4; ++nt)
        acc[mt][nt] = WMMA_BF16(af[mt], bfr[nt], acc[mt][nt]);

    __syncthreads();         // protect buf cur before it is refilled next step
  }

#pragma unroll
  for (int mt = 0; mt < 2; ++mt) {
#pragma unroll
    for (int nt = 0; nt < 4; ++nt) {
      const int col = nBlock + wn * 64 + nt * 16 + lm;
      const float bv = bias[col];
      if (MODE == 0) {
        const float qscale = 0.08838834764831845f;  // 1/sqrt(128)
        const int type = col >> 11;                 // 0=q 1=k 2=v (uniform per tile)
        const int h = (col >> 7) & (H - 1);
        const int d = col & (HD - 1);
#pragma unroll
        for (int r = 0; r < 8; ++r) {
          const int row = mBlock + wm * 32 + mt * 16 + half * 8 + r;
          const int b = row >> 11, s = row & (S - 1);
          const float v = acc[mt][nt][r] + bv;
          const long long bh = (long long)(b * H + h);
          if (type == 0)      Qh[(bh * S + s) * HD + d] = (__bf16)(v * qscale);
          else if (type == 1) Kh[(bh * S + s) * HD + d] = (__bf16)v;
          else                Vt[(bh * HD + d) * S + s] = (__bf16)v;
        }
      } else {
#pragma unroll
        for (int r = 0; r < 8; ++r) {
          const int row = mBlock + wm * 32 + mt * 16 + half * 8 + r;
          FO[(long long)row * D + col] = acc[mt][nt][r] + bv;
        }
      }
    }
  }
}

// ---------------- fused causal ALiBi flash attention ----------------
// grid = (S/64, B*H); 4 waves/block, each wave owns a 16-query tile.
__global__ void __launch_bounds__(128)
attn_kernel(const __bf16* __restrict__ Qh, const __bf16* __restrict__ Kh,
            const __bf16* __restrict__ Vt, __bf16* __restrict__ O) {
  __shared__ __align__(32) __bf16 pbuf[4][16 * 32];
  const int wave = threadIdx.x >> 5, lane = threadIdx.x & 31;
  const int half = lane >> 4, lm = lane & 15;
  const int bh = blockIdx.y;
  const int h = bh & (H - 1);
  const int qBase = blockIdx.x * 64 + wave * 16;
  const float slope = exp2f(-0.5f * (float)(h + 1));  // H=16 ALiBi slopes

  const __bf16* qp = Qh + ((long long)bh * S + qBase + lm) * HD;
  v16bf qf[4];
#pragma unroll
  for (int c = 0; c < 4; ++c) {
    v8bf a0 = *(const v8bf*)(qp + c * 32 + half * 8);
    v8bf a1 = *(const v8bf*)(qp + c * 32 + 16 + half * 8);
    qf[c] = cat16(a0, a1);
  }

  float m[8], l[8];
  v8f acc[8];
#pragma unroll
  for (int r = 0; r < 8; ++r) { m[r] = -3.0e38f; l[r] = 0.0f; }
#pragma unroll
  for (int n = 0; n < 8; ++n) acc[n] = v8f{};

  const int kEnd = qBase + 16;  // causal bound (exclusive)
  for (int kBase = 0; kBase < kEnd; kBase += 32) {
    if (kBase + 32 < kEnd) {
      __builtin_prefetch(Kh + ((long long)bh * S + kBase + 32 + lm) * HD, 0, 3);
      __builtin_prefetch(Vt + ((long long)bh * HD + lm) * S + kBase + 32, 0, 3);
    }
    v8f sc[2];
#pragma unroll
    for (int t = 0; t < 2; ++t) {
      const __bf16* kp = Kh + ((long long)bh * S + kBase + t * 16 + lm) * HD + half * 8;
      v8f s = v8f{};
#pragma unroll
      for (int c = 0; c < 4; ++c) {
        v16bf kf = cat16(*(const v8bf*)(kp + c * 32), *(const v8bf*)(kp + c * 32 + 16));
        s = WMMA_BF16(qf[c], kf, s);
      }
      sc[t] = s;
    }

    const int key0 = kBase + lm, key1 = kBase + 16 + lm;
    const float al0 = slope * (float)(key0 - (S - 1));
    const float al1 = slope * (float)(key1 - (S - 1));
    float p0[8], p1[8], alpha[8];
#pragma unroll
    for (int r = 0; r < 8; ++r) {
      int q = qBase + half * 8 + r;
      float s0 = sc[0][r] + al0; if (key0 > q) s0 = -3.0e38f;
      float s1 = sc[1][r] + al1; if (key1 > q) s1 = -3.0e38f;
      float mt = fmaxf(s0, s1);
#pragma unroll
      for (int off = 1; off < 16; off <<= 1)
        mt = fmaxf(mt, __shfl_xor(mt, off, 32));  // row reduce within half-wave
      float mnew = fmaxf(m[r], mt);
      float a = __expf(m[r] - mnew);
      float e0 = __expf(s0 - mnew);
      float e1 = __expf(s1 - mnew);
      float rs = e0 + e1;
#pragma unroll
      for (int off = 1; off < 16; off <<= 1)
        rs += __shfl_xor(rs, off, 32);
      l[r] = l[r] * a + rs;
      m[r] = mnew;
      alpha[r] = a; p0[r] = e0; p1[r] = e1;
    }
#pragma unroll
    for (int n = 0; n < 8; ++n) {
#pragma unroll
      for (int r = 0; r < 8; ++r) acc[n][r] *= alpha[r];
    }

    // C-layout P -> A-layout via per-wave LDS tile
    __bf16* pb = pbuf[wave];
#pragma unroll
    for (int r = 0; r < 8; ++r) {
      pb[(half * 8 + r) * 32 + lm]      = (__bf16)p0[r];
      pb[(half * 8 + r) * 32 + 16 + lm] = (__bf16)p1[r];
    }
    __builtin_amdgcn_wave_barrier();
    asm volatile("s_wait_dscnt 0" ::: "memory");  // cross-lane LDS visibility within wave
    v8bf pa0 = *(const v8bf*)(pb + lm * 32 + half * 8);
    v8bf pa1 = *(const v8bf*)(pb + lm * 32 + 16 + half * 8);
    v16bf pf = cat16(pa0, pa1);

    const __bf16* vp = Vt + ((long long)bh * HD + lm) * S + kBase + half * 8;
#pragma unroll
    for (int n = 0; n < 8; ++n) {
      v16bf vf = cat16(*(const v8bf*)(vp + (long long)n * 16 * S), *(const v8bf*)(vp + (long long)n * 16 * S + 16));
      acc[n] = WMMA_BF16(pf, vf, acc[n]);
    }
    __builtin_amdgcn_wave_barrier();
  }

  const int b = bh >> 4;
#pragma unroll
  for (int r = 0; r < 8; ++r) {
    int s = qBase + half * 8 + r;
    float inv = 1.0f / l[r];
    __bf16* op = O + ((long long)(b * S + s)) * D + h * HD;
#pragma unroll
    for (int n = 0; n < 8; ++n) op[n * 16 + lm] = (__bf16)(acc[n][r] * inv);
  }
}

extern "C" void kernel_launch(void* const* d_in, const int* in_sizes, int n_in,
                              void* d_out, int out_size, void* d_ws, size_t ws_size,
                              hipStream_t stream) {
  const float* x    = (const float*)d_in[0];
  const float* Wqkv = (const float*)d_in[1];
  const float* bqkv = (const float*)d_in[2];
  const float* Wout = (const float*)d_in[3];
  const float* bout = (const float*)d_in[4];
  float* out = (float*)d_out;

  char* ws = (char*)d_ws;
  size_t off = 0;
  auto alloc = [&](size_t bytes) {
    char* p = ws + off;
    off += (bytes + 255) & ~(size_t)255;
    return (void*)p;
  };
  __bf16* xb  = (__bf16*)alloc((size_t)M1 * K1 * 2);  // 32 MB
  __bf16* wqT = (__bf16*)alloc((size_t)N1 * K1 * 2);  // 24 MB
  __bf16* woT = (__bf16*)alloc((size_t)D * D * 2);    //  8 MB
  __bf16* Qh  = (__bf16*)alloc((size_t)M1 * D * 2);   // 32 MB
  __bf16* Kh  = (__bf16*)alloc((size_t)M1 * D * 2);   // 32 MB
  __bf16* Vt  = (__bf16*)alloc((size_t)M1 * D * 2);   // 32 MB
  __bf16* AO  = (__bf16*)alloc((size_t)M1 * D * 2);   // 32 MB

  {
    int n = M1 * K1;
    cvt_bf16_kernel<<<(n + 255) / 256, 256, 0, stream>>>(x, xb, n);
  }
  transpose_bf16_kernel<<<dim3(N1 / 32, K1 / 32), 256, 0, stream>>>(Wqkv, wqT, K1, N1);
  transpose_bf16_kernel<<<dim3(D / 32, D / 32), 256, 0, stream>>>(Wout, woT, D, D);

  gemm_bf16_kernel<0><<<dim3(N1 / 128, M1 / 128), 256, 0, stream>>>(
      xb, wqT, bqkv, Qh, Kh, Vt, nullptr);
  attn_kernel<<<dim3(S / 64, Bb * H), 128, 0, stream>>>(Qh, Kh, Vt, AO);
  gemm_bf16_kernel<1><<<dim3(D / 128, M1 / 128), 256, 0, stream>>>(
      AO, woT, bout, nullptr, nullptr, nullptr, out);
}
